// ResidualGatedGCNModel_74277164417322
// MI455X (gfx1250) — matrix operand order repaired and checked by reading, not compiled, not run on hardware
//
#include <hip/hip_runtime.h>
#include <hip/hip_bf16.h>

#define NN 2000
#define EE 32000
#define HH 300
#define HP 320           // padded width (K and N pad for WMMA)
#define LL 30
#define KT 10            // HP/32 K-tiles
#define NT 20            // HP/16 N-tiles
#define PACK_TILE 512    // 32 lanes * 16 bf16 per 16x16x32 B fragment
#define PACK_MAT (KT*NT*PACK_TILE)   // 102400 elements per packed HxH matrix

typedef __attribute__((ext_vector_type(16))) __bf16 v16bf;
typedef __attribute__((ext_vector_type(8)))  __bf16 v8bf;
typedef __attribute__((ext_vector_type(8)))  float  v8f;

__device__ __forceinline__ unsigned short f2bf(float f) {
  unsigned int u = __float_as_uint(f);
  u = u + 0x7FFFu + ((u >> 16) & 1u);      // round-to-nearest-even
  return (unsigned short)(u >> 16);
}
__device__ __forceinline__ float sigmoidf_(float x) { return 1.f / (1.f + expf(-x)); }

// ---- pack fp32 [L,H,H] weights into per-lane bf16 B-fragment layout ----
// layout: [(l*4+mat)][kt][nt][lane][16], fragment: n = nt*16+(lane&15),
//         k = kt*32 + (lane>>4)*16 + e   (e = 0..15 contiguous per lane)
__global__ void pack_weights_k(const float* __restrict__ eU, const float* __restrict__ eW,
                               const float* __restrict__ nU, const float* __restrict__ nV,
                               unsigned short* __restrict__ wp) {
  const int blk  = blockIdx.x;                 // (l*4+mat)*(KT*NT) + kt*NT + nt
  const int tile = blk % (KT * NT);
  const int lm   = blk / (KT * NT);
  const int mat  = lm & 3, l = lm >> 2;
  const int kt = tile / NT, nt = tile % NT;
  const int tid = threadIdx.x;                 // 0..511
  const int lane = tid >> 4, e = tid & 15;
  const int n = nt * 16 + (lane & 15);
  const int k = kt * 32 + (lane >> 4) * 16 + e;
  const float* W = (mat == 0) ? eU : (mat == 1) ? eW : (mat == 2) ? nU : nV;
  float v = (k < HH && n < HH) ? W[((size_t)l * HH + k) * HH + n] : 0.f;
  wp[(size_t)blk * PACK_TILE + tid] = f2bf(v);
}

// ---- bf16 WMMA GEMM: C[M,HP] = A[M,HP](bf16,row-major) * Bpacked ----
// grid: (M/16, 5) ; one wave computes a 16x64 output tile (4 N-tiles)
__global__ void __launch_bounds__(32) gemm_bf16(const unsigned short* __restrict__ A,
                                                const unsigned short* __restrict__ Bp,
                                                float* __restrict__ C) {
  const int lane = threadIdx.x;
  const int mt = blockIdx.x;
  const int ng = blockIdx.y;                    // group of 4 N-tiles
  const int half = lane >> 4;
  const int row = mt * 16 + (lane & 15);
  const unsigned short* arow = A + (size_t)row * HP;
  v8f a0 = {0.f,0.f,0.f,0.f,0.f,0.f,0.f,0.f};
  v8f a1 = a0, a2 = a0, a3 = a0;
  for (int kt = 0; kt < KT; ++kt) {
    const int k0 = kt * 32 + half * 16;
    // 16-bit A-fragment: K in [k0,k0+16) contiguous, matching packed B
    v8bf lo = *(const v8bf*)(arow + k0);
    v8bf hi = *(const v8bf*)(arow + k0 + 8);
    v16bf a = __builtin_shufflevector(lo, hi, 0,1,2,3,4,5,6,7,8,9,10,11,12,13,14,15);
    const unsigned short* bt = Bp + ((size_t)(kt * NT + ng * 4) * PACK_TILE) + lane * 16;
    v16bf b0 = *(const v16bf*)(bt);
    v16bf b1 = *(const v16bf*)(bt + PACK_TILE);
    v16bf b2 = *(const v16bf*)(bt + 2 * PACK_TILE);
    v16bf b3 = *(const v16bf*)(bt + 3 * PACK_TILE);
    a0 = __builtin_amdgcn_wmma_f32_16x16x32_bf16(false, a, false, b0, (short)0, a0, false, false);
    a1 = __builtin_amdgcn_wmma_f32_16x16x32_bf16(false, a, false, b1, (short)0, a1, false, false);
    a2 = __builtin_amdgcn_wmma_f32_16x16x32_bf16(false, a, false, b2, (short)0, a2, false, false);
    a3 = __builtin_amdgcn_wmma_f32_16x16x32_bf16(false, a, false, b3, (short)0, a3, false, false);
  }
  // D layout: lane n = lane&15 ; VGPR r -> m = r + 8*(lane>>4)
  const int mbase = mt * 16 + half * 8;
  const int ncol = (lane & 15);
  #pragma unroll
  for (int r = 0; r < 8; ++r) {
    float* crow = C + (size_t)(mbase + r) * HP;
    crow[(ng * 4 + 0) * 16 + ncol] = a0[r];
    crow[(ng * 4 + 1) * 16 + ncol] = a1[r];
    crow[(ng * 4 + 2) * 16 + ncol] = a2[r];
    crow[(ng * 4 + 3) * 16 + ncol] = a3[r];
  }
}

// ---- elementwise / reduction kernels (blockDim = 320, thread = column) ----
__global__ void node_emb_k(const float* __restrict__ x, const float* __restrict__ W,
                           const float* __restrict__ b, float* __restrict__ out) {
  int n = blockIdx.x, c = threadIdx.x;
  if (c >= HH) return;
  float s = b[c];
  #pragma unroll
  for (int i = 0; i < 4; ++i) s += x[n * 4 + i] * W[i * HH + c];
  out[(size_t)n * HP + c] = s;
}

__global__ void edge_emb_k(const float* __restrict__ ea, const float* __restrict__ dW,
                           const float* __restrict__ db, const float* __restrict__ tW,
                           const float* __restrict__ tb, float* __restrict__ out) {
  int e = blockIdx.x, c = threadIdx.x;
  if (c >= HH) return;
  float v = (c < 150) ? (ea[e * 2 + 0] * dW[c] + db[c])
                      : (ea[e * 2 + 1] * tW[c - 150] + tb[c - 150]);
  out[(size_t)e * HP + c] = v;
}

__global__ void conv_bf_k(const float* __restrict__ in, unsigned short* __restrict__ out) {
  int r = blockIdx.x, c = threadIdx.x;
  out[(size_t)r * HP + c] = (c < HH) ? f2bf(in[(size_t)r * HP + c]) : (unsigned short)0;
}

__global__ void zero_f32_k(float* p, size_t n) {
  size_t i = (size_t)blockIdx.x * blockDim.x + threadIdx.x;
  size_t stride = (size_t)gridDim.x * blockDim.x;
  for (; i < n; i += stride) p[i] = 0.f;
}

__global__ void edge_assemble_k(float* __restrict__ gate, const float* __restrict__ nodeU,
                                const int* __restrict__ ei, const float* __restrict__ eWb,
                                const float* __restrict__ eUb) {
  int e = blockIdx.x, c = threadIdx.x;
  if (c >= HH) return;
  int s = ei[e], d = ei[EE + e];
  gate[(size_t)e * HP + c] += eWb[c] + 2.f * eUb[c]
                            + nodeU[(size_t)s * HP + c] + nodeU[(size_t)d * HP + c];
}

__global__ void reduce_cols_k(const float* __restrict__ src, int rows, int chunk,
                              float* __restrict__ osum, float* __restrict__ osq) {
  int c = threadIdx.x;
  if (c >= HH) return;
  int r0 = blockIdx.x * chunk, r1 = r0 + chunk;
  if (r1 > rows) r1 = rows;
  float s = 0.f, sq = 0.f;
  for (int r = r0; r < r1; ++r) { float v = src[(size_t)r * HP + c]; s += v; sq += v * v; }
  atomicAdd(&osum[c], s);
  atomicAdd(&osq[c], sq);
}

__global__ void sig_colsum_k(const float* __restrict__ edge, int rows, int chunk,
                             float* __restrict__ osum) {
  int c = threadIdx.x;
  if (c >= HH) return;
  int r0 = blockIdx.x * chunk, r1 = r0 + chunk;
  if (r1 > rows) r1 = rows;
  float s = 0.f;
  for (int r = r0; r < r1; ++r) s += sigmoidf_(edge[(size_t)r * HP + c]);
  atomicAdd(&osum[c], s);
}

__global__ void msg_scatter_k(const float* __restrict__ edge, const float* __restrict__ nodeV,
                              const float* __restrict__ sigsum, const float* __restrict__ nVb,
                              const int* __restrict__ ei, float* __restrict__ agg) {
  int e = blockIdx.x, c = threadIdx.x;
  if (c >= HH) return;
  float sig = sigmoidf_(edge[(size_t)e * HP + c]);
  float eta = sig / (sigsum[c] + 1e-20f);
  int s = ei[e], d = ei[EE + e];
  atomicAdd(&agg[(size_t)d * HP + c], eta * (nodeV[(size_t)s * HP + c] + nVb[c]));
}

__global__ void node_assemble_k(float* __restrict__ gate, const float* __restrict__ agg,
                                const float* __restrict__ nUb) {
  int n = blockIdx.x, c = threadIdx.x;
  if (c >= HH) return;
  gate[(size_t)n * HP + c] += nUb[c] + agg[(size_t)n * HP + c];
}

// BN(training stats) + ReLU + residual, shared by edge/node paths
__global__ void bn_update_k(const float* __restrict__ in, const float* __restrict__ gate,
                            const float* __restrict__ csum, const float* __restrict__ csq,
                            const float* __restrict__ g, const float* __restrict__ b,
                            float inv_rows, float* __restrict__ out) {
  int r = blockIdx.x, c = threadIdx.x;
  if (c >= HH) return;
  float mean = csum[c] * inv_rows;
  float var  = csq[c] * inv_rows - mean * mean;
  float v = (gate[(size_t)r * HP + c] - mean) * rsqrtf(var + 1e-5f) * g[c] + b[c];
  out[(size_t)r * HP + c] = in[(size_t)r * HP + c] + fmaxf(v, 0.f);
}

// wave-per-edge classifier: dot(edge[e], clsW) -> sigmoid -> scatter + BCE
__global__ void classify_k(const float* __restrict__ edge, const int* __restrict__ ei,
                           const float* __restrict__ y, const float* __restrict__ clsW,
                           const float* __restrict__ clsb, float* __restrict__ pred_adj,
                           float* __restrict__ lossacc) {
  int gid = blockIdx.x * blockDim.x + threadIdx.x;
  int e = gid >> 5, lane = gid & 31;
  if (e >= EE) return;
  float s = 0.f;
  for (int c = lane; c < HH; c += 32) s += edge[(size_t)e * HP + c] * clsW[c];
  #pragma unroll
  for (int off = 16; off > 0; off >>= 1) s += __shfl_down(s, off, 32);
  if (lane == 0) {
    float p = sigmoidf_(s + clsb[0]);
    int src = ei[e], dst = ei[EE + e];
    atomicAdd(&pred_adj[(size_t)src * NN + dst], p);
    float pc = fminf(fmaxf(p, 1e-7f), 1.f - 1e-7f);
    float li = -(y[e] * logf(pc) + (1.f - y[e]) * log1pf(-pc));
    atomicAdd(lossacc, li);
  }
}

__global__ void finalize_k(const float* __restrict__ lossacc, float* __restrict__ out) {
  out[(size_t)NN * NN] = lossacc[0] * (1.f / (float)EE);
}

extern "C" void kernel_launch(void* const* d_in, const int* in_sizes, int n_in,
                              void* d_out, int out_size, void* d_ws, size_t ws_size,
                              hipStream_t stream) {
  const float* x   = (const float*)d_in[0];
  const float* ea  = (const float*)d_in[1];
  const int*   ei  = (const int*)d_in[2];
  const float* y   = (const float*)d_in[3];
  const float* nW  = (const float*)d_in[4];
  const float* nb  = (const float*)d_in[5];
  const float* dW  = (const float*)d_in[6];
  const float* db  = (const float*)d_in[7];
  const float* tW  = (const float*)d_in[8];
  const float* tb  = (const float*)d_in[9];
  const float* eUW = (const float*)d_in[10];
  const float* eUb = (const float*)d_in[11];
  const float* eWW = (const float*)d_in[12];
  const float* eWb = (const float*)d_in[13];
  const float* nUW = (const float*)d_in[14];
  const float* nUb = (const float*)d_in[15];
  const float* nVW = (const float*)d_in[16];
  const float* nVb = (const float*)d_in[17];
  const float* beg = (const float*)d_in[18];
  const float* beb = (const float*)d_in[19];
  const float* bng = (const float*)d_in[20];
  const float* bnb = (const float*)d_in[21];
  const float* cW  = (const float*)d_in[22];
  const float* cb  = (const float*)d_in[23];
  float* out = (float*)d_out;
  (void)in_sizes; (void)n_in; (void)out_size; (void)ws_size;

  char* wsp = (char*)d_ws;
  auto alloc = [&](size_t bytes) -> void* {
    void* p = (void*)wsp;
    wsp += (bytes + 255) & ~(size_t)255;
    return p;
  };
  unsigned short* WP     = (unsigned short*)alloc((size_t)LL * 4 * PACK_MAT * 2);
  float* nodeA           = (float*)alloc((size_t)NN * HP * 4);
  float* nodeB           = (float*)alloc((size_t)NN * HP * 4);
  float* edgeA           = (float*)alloc((size_t)EE * HP * 4);
  float* edgeB           = (float*)alloc((size_t)EE * HP * 4);
  unsigned short* nodeBF = (unsigned short*)alloc((size_t)NN * HP * 2);
  unsigned short* edgeBF = (unsigned short*)alloc((size_t)EE * HP * 2);
  float* gateE           = (float*)alloc((size_t)EE * HP * 4);
  float* gateN           = (float*)alloc((size_t)NN * HP * 4);
  float* nodeU           = (float*)alloc((size_t)NN * HP * 4);
  float* nodeV           = (float*)alloc((size_t)NN * HP * 4);
  float* agg             = (float*)alloc((size_t)NN * HP * 4);
  float* stats           = (float*)alloc(2048 * 4);
  float* ecsum = stats,        *ecsq = stats + HP,  *sigsum = stats + 2 * HP;
  float* ncsum = stats + 3*HP, *ncsq = stats + 4*HP, *lossacc = stats + 5 * HP;

  dim3 b320(HP);
  pack_weights_k<<<LL * 4 * KT * NT, PACK_TILE, 0, stream>>>(eUW, eWW, nUW, nVW, WP);
  node_emb_k<<<NN, b320, 0, stream>>>(x, nW, nb, nodeA);
  edge_emb_k<<<EE, b320, 0, stream>>>(ea, dW, db, tW, tb, edgeA);

  float* ncur = nodeA; float* nnxt = nodeB;
  float* ecur = edgeA; float* enxt = edgeB;
  for (int l = 0; l < LL; ++l) {
    conv_bf_k<<<NN, b320, 0, stream>>>(ncur, nodeBF);
    conv_bf_k<<<EE, b320, 0, stream>>>(ecur, edgeBF);
    zero_f32_k<<<1024, 256, 0, stream>>>(agg, (size_t)NN * HP);
    zero_f32_k<<<4, 256, 0, stream>>>(stats, 2048);
    const unsigned short* wp_eU = WP + (size_t)(l * 4 + 0) * PACK_MAT;
    const unsigned short* wp_eW = WP + (size_t)(l * 4 + 1) * PACK_MAT;
    const unsigned short* wp_nU = WP + (size_t)(l * 4 + 2) * PACK_MAT;
    const unsigned short* wp_nV = WP + (size_t)(l * 4 + 3) * PACK_MAT;
    gemm_bf16<<<dim3(NN / 16, 5), 32, 0, stream>>>(nodeBF, wp_eU, nodeU);  // node @ eUW
    gemm_bf16<<<dim3(EE / 16, 5), 32, 0, stream>>>(edgeBF, wp_eW, gateE);  // edge @ eWW
    gemm_bf16<<<dim3(NN / 16, 5), 32, 0, stream>>>(nodeBF, wp_nV, nodeV);  // node @ nVW
    gemm_bf16<<<dim3(NN / 16, 5), 32, 0, stream>>>(nodeBF, wp_nU, gateN);  // node @ nUW
    edge_assemble_k<<<EE, b320, 0, stream>>>(gateE, nodeU, ei, eWb + l * HH, eUb + l * HH);
    reduce_cols_k<<<125, b320, 0, stream>>>(gateE, EE, 256, ecsum, ecsq);
    sig_colsum_k<<<125, b320, 0, stream>>>(ecur, EE, 256, sigsum);
    msg_scatter_k<<<EE, b320, 0, stream>>>(ecur, nodeV, sigsum, nVb + l * HH, ei, agg);
    bn_update_k<<<EE, b320, 0, stream>>>(ecur, gateE, ecsum, ecsq,
                                         beg + l * HH, beb + l * HH, 1.f / EE, enxt);
    node_assemble_k<<<NN, b320, 0, stream>>>(gateN, agg, nUb + l * HH);
    reduce_cols_k<<<25, b320, 0, stream>>>(gateN, NN, 80, ncsum, ncsq);
    bn_update_k<<<NN, b320, 0, stream>>>(ncur, gateN, ncsum, ncsq,
                                         bng + l * HH, bnb + l * HH, 1.f / NN, nnxt);
    float* t;
    t = ncur; ncur = nnxt; nnxt = t;
    t = ecur; ecur = enxt; enxt = t;
  }

  zero_f32_k<<<4096, 256, 0, stream>>>(out, (size_t)NN * NN + 1);
  zero_f32_k<<<1, 32, 0, stream>>>(lossacc, 1);
  classify_k<<<(EE * 32 + 255) / 256, 256, 0, stream>>>(ecur, ei, y, cW, cb, out, lossacc);
  finalize_k<<<1, 1, 0, stream>>>(lossacc, out);
}
